// Seq2Seq_66924180406281
// MI455X (gfx1250) — matrix operand order repaired and checked by reading, not compiled, not run on hardware
//
#include <hip/hip_runtime.h>
#include <hip/hip_fp16.h>
#include <stdint.h>

typedef _Float16 half_t;
typedef __attribute__((ext_vector_type(16))) _Float16 v16h;
typedef __attribute__((ext_vector_type(8)))  float    v8f;

#define VOCAB 32000
#define EDIM  512
#define HDIM  512
#define NB    32
#define LSRC  128
#define LTGT  48
#define G4H   (4 * HDIM)

__device__ __forceinline__ float sigmoidf_(float x) { return 1.0f / (1.0f + __expf(-x)); }

// ---------------------------------------------------------------------------
// Generic WMMA GEMM:  C[M x Nc] = A[M x K] (f16, row-major) * W[Nc x K]^T (f16)
//                     + b1 + b2, optional tanh; write f32 and/or f16 outputs.
// One 16x16 output tile per wave, 8 waves per block.
// Fragment layouts: contiguous k, lane half h holds k = 16h..16h+15 of A and B (wave32, 16x32 / 32x16).
// ---------------------------------------------------------------------------
__global__ __launch_bounds__(256) void k_gemm(
    const half_t* __restrict__ A, int lda,
    const half_t* __restrict__ W, int ldw,
    const float* __restrict__ b1, const float* __restrict__ b2,
    float* __restrict__ outF, long ldoF,
    half_t* __restrict__ outH, long ldoH,
    int M, int Nc, int K, int act)
{
  const int wave = threadIdx.x >> 5;
  const int lane = threadIdx.x & 31;
  const int tile = blockIdx.x * 8 + wave;
  const int mTiles = M >> 4;
  const int total = mTiles * (Nc >> 4);
  if (tile >= total) return;            // whole-wave uniform exit (EXEC all-1s for WMMA)
  const int mt = tile % mTiles;
  const int nt = tile / mTiles;

  // A: lane<16 -> row m0+lane holds K = 0..15; lane>=16 -> K = 16..31
  const half_t* Ap = A + (long)(mt * 16 + (lane & 15)) * lda + ((lane >> 4) << 4);
  // B from W row-major (Nc x K): lane&15 = column n, lane>=16 covers K 16..31
  const half_t* Wp = W + (long)(nt * 16 + (lane & 15)) * ldw + ((lane >> 4) << 4);

  v8f acc = {};
  for (int k0 = 0; k0 < K; k0 += 32) {
    v16h a, b;
#pragma unroll
    for (int i = 0; i < 8; ++i) { a[i] = Ap[i]; a[i + 8] = Ap[8 + i]; }
#pragma unroll
    for (int i = 0; i < 16; ++i) b[i] = Wp[i];
    acc = __builtin_amdgcn_wmma_f32_16x16x32_f16(false, a, false, b, (short)0, acc,
                                                 false, false);
    Ap += 32; Wp += 32;
  }

  const int n = nt * 16 + (lane & 15);
  float bias = (b1 ? b1[n] : 0.0f) + (b2 ? b2[n] : 0.0f);
#pragma unroll
  for (int j = 0; j < 8; ++j) {
    const int r = mt * 16 + j + ((lane >> 4) << 3);   // C layout: VGPR j -> M=j / M=8+j
    float v = acc[j] + bias;
    if (act == 1) v = tanhf(v);
    if (outF) outF[(long)r * ldoF + n] = v;
    if (outH) outH[(long)r * ldoH + n] = (half_t)v;
  }
}

// ---------------------------------------------------------------------------
// Fused LSTM recurrent step.  gridDim = (HDIM/16, nDirs), 256 threads.
// Each block: one 16-wide hidden slice for one direction.
// 8 waves = (2 batch tiles) x (4 gates i,f,g,o); each wave does WMMA K-loop
// over Whh (and optionally a second input GEMM for decoder layer 1),
// stages gates in LDS, then the block applies the nonlinearity.
// ---------------------------------------------------------------------------
struct StepArgs {
  const float*  pg[2];     // pre-gates for this timestep [M x 4H] (x@Wih^T + biases), or null
  const half_t* x[2];      // optional dynamic input [M x Kx]
  const half_t* Wx[2];     // [4H x Kx]
  int           Kx;
  const half_t* Whh[2];    // [4H x H]
  const half_t* hIn[2];    // [M x H] f16
  half_t*       hOut[2];   // [M x H] f16
  float*        c[2];      // [M x H] f32, read+write
  const float*  b1[2];     // optional bias (when pg == null)
  const float*  b2[2];
  half_t*       y[2];      // optional: also write h into y (row = batch), or null
  long          ldy;
  int           yoff[2];
};

__global__ __launch_bounds__(256) void k_lstm_step(StepArgs s) {
  const int d    = blockIdx.y;
  const int j    = blockIdx.x;            // hidden tile (16 units)
  const int wave = threadIdx.x >> 5;
  const int lane = threadIdx.x & 31;
  const int gate = wave & 3;
  const int mt   = wave >> 2;

  __shared__ float gbuf[8][16][16];

  v8f acc = {};
  {
    const half_t* Wp = s.Whh[d] + (long)(gate * HDIM + j * 16 + (lane & 15)) * HDIM
                                + ((lane >> 4) << 4);
    const half_t* Ap = s.hIn[d] + (long)(mt * 16 + (lane & 15)) * HDIM
                                + ((lane >> 4) << 4);
    for (int k0 = 0; k0 < HDIM; k0 += 32) {
      v16h a, b;
#pragma unroll
      for (int i = 0; i < 8; ++i) { a[i] = Ap[i]; a[i + 8] = Ap[8 + i]; }
#pragma unroll
      for (int i = 0; i < 16; ++i) b[i] = Wp[i];
      acc = __builtin_amdgcn_wmma_f32_16x16x32_f16(false, a, false, b, (short)0, acc,
                                                   false, false);
      Ap += 32; Wp += 32;
    }
  }
  if (s.x[d]) {  // decoder layer 1: g += h0 @ Wih^T  (uniform branch)
    const half_t* Wp = s.Wx[d] + (long)(gate * HDIM + j * 16 + (lane & 15)) * s.Kx
                               + ((lane >> 4) << 4);
    const half_t* Ap = s.x[d]  + (long)(mt * 16 + (lane & 15)) * s.Kx
                               + ((lane >> 4) << 4);
    for (int k0 = 0; k0 < s.Kx; k0 += 32) {
      v16h a, b;
#pragma unroll
      for (int i = 0; i < 8; ++i) { a[i] = Ap[i]; a[i + 8] = Ap[8 + i]; }
#pragma unroll
      for (int i = 0; i < 16; ++i) b[i] = Wp[i];
      acc = __builtin_amdgcn_wmma_f32_16x16x32_f16(false, a, false, b, (short)0, acc,
                                                   false, false);
      Ap += 32; Wp += 32;
    }
  }

  {
    const int n   = lane & 15;
    const int col = gate * HDIM + j * 16 + n;
    float bias = (s.b1[d] ? s.b1[d][col] : 0.0f) + (s.b2[d] ? s.b2[d][col] : 0.0f);
#pragma unroll
    for (int jr = 0; jr < 8; ++jr) {
      const int rloc = jr + ((lane >> 4) << 3);
      const int r    = mt * 16 + rloc;
      float v = acc[jr] + bias;
      if (s.pg[d]) v += s.pg[d][(long)r * G4H + col];
      gbuf[wave][rloc][n] = v;
    }
  }
  __syncthreads();

  // gating: 32 batch x 16 hidden = 512 elements, 2 per thread
  for (int idx = threadIdx.x; idx < NB * 16; idx += 256) {
    const int m   = idx >> 4;
    const int u   = idx & 15;
    const int mt2 = m >> 4;
    const int mr  = m & 15;
    const float gi = gbuf[mt2 * 4 + 0][mr][u];
    const float gf = gbuf[mt2 * 4 + 1][mr][u];
    const float gg = gbuf[mt2 * 4 + 2][mr][u];
    const float go = gbuf[mt2 * 4 + 3][mr][u];
    const long off = (long)m * HDIM + j * 16 + u;
    float cn = sigmoidf_(gf) * s.c[d][off] + sigmoidf_(gi) * tanhf(gg);
    s.c[d][off] = cn;
    const float h = sigmoidf_(go) * tanhf(cn);
    s.hOut[d][off] = (half_t)h;
    if (s.y[d]) s.y[d][(long)m * s.ldy + s.yoff[d] + j * 16 + u] = (half_t)h;
  }
}

// ---------------------------------------------------------------------------
// Small helper kernels
// ---------------------------------------------------------------------------
__global__ void k_cast(const float* __restrict__ s, half_t* __restrict__ d, long n) {
  long i = (long)blockIdx.x * 256 + threadIdx.x;
  if (i < n) d[i] = (half_t)s[i];
}

__global__ void k_zero(uint32_t* __restrict__ p, long n) {
  long i = (long)blockIdx.x * 256 + threadIdx.x;
  if (i < n) p[i] = 0u;
}

// embedding gather + f16 cast; rows ordered (t*NB + n)
__global__ void k_embed(const int* __restrict__ tok, int T, int sos_shift,
                        const float* __restrict__ emb, half_t* __restrict__ out) {
  const int r = blockIdx.x;         // r = t*NB + n
  const int t = r / NB;
  const int n = r % NB;
  int token;
  if (sos_shift) token = (t == 0) ? 1 : tok[n * T + (t - 1)];
  else           token = tok[n * T + t];
  const float* e = emb + (long)token * EDIM;
  for (int i = threadIdx.x; i < EDIM; i += blockDim.x)
    out[(long)r * EDIM + i] = (half_t)e[i];
}

// energy[n][s] = dot(proj[s*NB+n][:], h1[n][:]) ; one wave per (n,s)
__global__ __launch_bounds__(256) void k_energy(const half_t* __restrict__ proj,
                                                const half_t* __restrict__ h1,
                                                float* __restrict__ energy) {
  const int g = blockIdx.x * 8 + (threadIdx.x >> 5);
  const int lane = threadIdx.x & 31;
  if (g >= NB * LSRC) return;
  const int n = g / LSRC;
  const int sPos = g % LSRC;
  const half_t* p = proj + (long)(sPos * NB + n) * HDIM + lane * 16;
  const half_t* hh = h1 + (long)n * HDIM + lane * 16;
  float acc = 0.0f;
#pragma unroll
  for (int i = 0; i < 16; ++i) acc += (float)p[i] * (float)hh[i];
#pragma unroll
  for (int o = 16; o >= 1; o >>= 1) acc += __shfl_xor(acc, o, 32);
  if (lane == 0) energy[n * LSRC + sPos] = acc;
}

__global__ __launch_bounds__(LSRC) void k_softmax(const float* __restrict__ energy,
                                                  float* __restrict__ score) {
  const int n = blockIdx.x;
  const int sPos = threadIdx.x;
  __shared__ float sh[LSRC];
  const float v = energy[n * LSRC + sPos];
  sh[sPos] = v; __syncthreads();
  for (int o = LSRC / 2; o >= 1; o >>= 1) {
    if (sPos < o) sh[sPos] = fmaxf(sh[sPos], sh[sPos + o]);
    __syncthreads();
  }
  const float mx = sh[0]; __syncthreads();
  const float e = __expf(v - mx);
  sh[sPos] = e; __syncthreads();
  for (int o = LSRC / 2; o >= 1; o >>= 1) {
    if (sPos < o) sh[sPos] += sh[sPos + o];
    __syncthreads();
  }
  score[n * LSRC + sPos] = e / sh[0];
}

// cat[n] = [ h1[n] (H) | ctx[n] (2H) ] in f16; ctx = sum_s score*enc
__global__ __launch_bounds__(256) void k_ctx(const float* __restrict__ score,
                                             const half_t* __restrict__ enc,
                                             const half_t* __restrict__ h1,
                                             half_t* __restrict__ cat) {
  const int n = blockIdx.x;
  for (int e = threadIdx.x; e < 2 * HDIM; e += 256) {
    float acc = 0.0f;
    for (int sPos = 0; sPos < LSRC; ++sPos)
      acc += score[n * LSRC + sPos] * (float)enc[(long)(sPos * NB + n) * (2 * HDIM) + e];
    cat[(long)n * (3 * HDIM) + HDIM + e] = (half_t)acc;
  }
  for (int i = threadIdx.x; i < HDIM; i += 256)
    cat[(long)n * (3 * HDIM) + i] = h1[(long)n * HDIM + i];
}

// ---------------------------------------------------------------------------
// Host orchestration
// ---------------------------------------------------------------------------
static inline int gemm_blocks(int M, int Nc) { return ((M >> 4) * (Nc >> 4) + 7) / 8; }

extern "C" void kernel_launch(void* const* d_in, const int* in_sizes, int n_in,
                              void* d_out, int out_size, void* d_ws, size_t ws_size,
                              hipStream_t stream) {
  (void)in_sizes; (void)n_in; (void)out_size; (void)ws_size;
  // ---- inputs (setup_inputs flattening order) ----
  const int*   src   = (const int*)d_in[0];
  const int*   tgt   = (const int*)d_in[1];
  const float* emb   = (const float*)d_in[2];
  const float* fWihF0 = (const float*)d_in[3];  const float* fWhhF0 = (const float*)d_in[4];
  const float* bihF0  = (const float*)d_in[5];  const float* bhhF0  = (const float*)d_in[6];
  const float* fWihB0 = (const float*)d_in[7];  const float* fWhhB0 = (const float*)d_in[8];
  const float* bihB0  = (const float*)d_in[9];  const float* bhhB0  = (const float*)d_in[10];
  const float* fWihF1 = (const float*)d_in[11]; const float* fWhhF1 = (const float*)d_in[12];
  const float* bihF1  = (const float*)d_in[13]; const float* bhhF1  = (const float*)d_in[14];
  const float* fWihB1 = (const float*)d_in[15]; const float* fWhhB1 = (const float*)d_in[16];
  const float* bihB1  = (const float*)d_in[17]; const float* bhhB1  = (const float*)d_in[18];
  const float* fdWih0 = (const float*)d_in[19]; const float* fdWhh0 = (const float*)d_in[20];
  const float* dbih0  = (const float*)d_in[21]; const float* dbhh0  = (const float*)d_in[22];
  const float* fdWih1 = (const float*)d_in[23]; const float* fdWhh1 = (const float*)d_in[24];
  const float* dbih1  = (const float*)d_in[25]; const float* dbhh1  = (const float*)d_in[26];
  const float* fattW  = (const float*)d_in[27]; const float* att_b  = (const float*)d_in[28];
  const float* flin1W = (const float*)d_in[29]; const float* lin1_b = (const float*)d_in[30];
  const float* flin2W = (const float*)d_in[31]; const float* lin2_b = (const float*)d_in[32];
  const float* init_c = (const float*)d_in[33];
  float* out = (float*)d_out;

  // ---- workspace carve ----
  char* wp = (char*)d_ws;
  auto alloc = [&](size_t bytes) -> void* {
    void* r = (void*)wp; wp += (bytes + 255) & ~(size_t)255; return r;
  };
  auto allocH = [&](long n) { return (half_t*)alloc((size_t)n * sizeof(half_t)); };
  auto allocF = [&](long n) { return (float*)alloc((size_t)n * sizeof(float)); };

  half_t* eWihF0 = allocH((long)G4H * EDIM);      half_t* eWhhF0 = allocH((long)G4H * HDIM);
  half_t* eWihB0 = allocH((long)G4H * EDIM);      half_t* eWhhB0 = allocH((long)G4H * HDIM);
  half_t* eWihF1 = allocH((long)G4H * 2 * HDIM);  half_t* eWhhF1 = allocH((long)G4H * HDIM);
  half_t* eWihB1 = allocH((long)G4H * 2 * HDIM);  half_t* eWhhB1 = allocH((long)G4H * HDIM);
  half_t* dWih0  = allocH((long)G4H * EDIM);      half_t* dWhh0  = allocH((long)G4H * HDIM);
  half_t* dWih1  = allocH((long)G4H * HDIM);      half_t* dWhh1  = allocH((long)G4H * HDIM);
  half_t* attW   = allocH((long)HDIM * 2 * HDIM);
  half_t* lin1W  = allocH((long)HDIM * 3 * HDIM);
  half_t* lin2W  = allocH((long)VOCAB * HDIM);

  half_t* srcX  = allocH((long)LSRC * NB * EDIM);
  float*  pgA   = allocF((long)LSRC * NB * G4H);
  float*  pgB   = allocF((long)LSRC * NB * G4H);
  half_t* y0    = allocH((long)LSRC * NB * 2 * HDIM);
  half_t* y1    = allocH((long)LSRC * NB * 2 * HDIM);
  half_t* proj  = allocH((long)LSRC * NB * HDIM);
  half_t* decX  = allocH((long)LTGT * NB * EDIM);
  float*  decPG = allocF((long)LTGT * NB * G4H);

  half_t* hF0[2] = { allocH(NB * HDIM), allocH(NB * HDIM) };
  half_t* hB0[2] = { allocH(NB * HDIM), allocH(NB * HDIM) };
  half_t* hF1[2] = { allocH(NB * HDIM), allocH(NB * HDIM) };
  half_t* hB1[2] = { allocH(NB * HDIM), allocH(NB * HDIM) };
  float*  cF0 = allocF(NB * HDIM); float* cB0 = allocF(NB * HDIM);
  float*  cF1 = allocF(NB * HDIM); float* cB1 = allocF(NB * HDIM);
  half_t* dh0[2] = { allocH(NB * HDIM), allocH(NB * HDIM) };
  half_t* dh1[2] = { allocH(NB * HDIM), allocH(NB * HDIM) };
  float*  dc0 = allocF(NB * HDIM); float* dc1 = allocF(NB * HDIM);
  float*  energy = allocF(NB * LSRC);
  float*  score  = allocF(NB * LSRC);
  half_t* cat    = allocH((long)NB * 3 * HDIM);
  half_t* hid    = allocH((long)NB * HDIM);

  auto cast = [&](const float* s, half_t* d, long n) {
    k_cast<<<(int)((n + 255) / 256), 256, 0, stream>>>(s, d, n);
  };
  auto zero = [&](void* p, long bytes) {
    long n = bytes / 4;
    k_zero<<<(int)((n + 255) / 256), 256, 0, stream>>>((uint32_t*)p, n);
  };

  // ---- 1. weight casts to f16 ----
  cast(fWihF0, eWihF0, (long)G4H * EDIM);     cast(fWhhF0, eWhhF0, (long)G4H * HDIM);
  cast(fWihB0, eWihB0, (long)G4H * EDIM);     cast(fWhhB0, eWhhB0, (long)G4H * HDIM);
  cast(fWihF1, eWihF1, (long)G4H * 2 * HDIM); cast(fWhhF1, eWhhF1, (long)G4H * HDIM);
  cast(fWihB1, eWihB1, (long)G4H * 2 * HDIM); cast(fWhhB1, eWhhB1, (long)G4H * HDIM);
  cast(fdWih0, dWih0, (long)G4H * EDIM);      cast(fdWhh0, dWhh0, (long)G4H * HDIM);
  cast(fdWih1, dWih1, (long)G4H * HDIM);      cast(fdWhh1, dWhh1, (long)G4H * HDIM);
  cast(fattW,  attW,  (long)HDIM * 2 * HDIM);
  cast(flin1W, lin1W, (long)HDIM * 3 * HDIM);
  cast(flin2W, lin2W, (long)VOCAB * HDIM);

  // ---- 2. state init ----
  zero(hF0[0], NB * HDIM * 2); zero(hF0[1], NB * HDIM * 2);
  zero(hB0[0], NB * HDIM * 2); zero(hB0[1], NB * HDIM * 2);
  zero(hF1[0], NB * HDIM * 2); zero(hF1[1], NB * HDIM * 2);
  zero(hB1[0], NB * HDIM * 2); zero(hB1[1], NB * HDIM * 2);
  zero(cF0, NB * HDIM * 4); zero(cB0, NB * HDIM * 4);
  zero(cF1, NB * HDIM * 4); zero(cB1, NB * HDIM * 4);

  // ---- 3. source embedding + encoder layer0 pre-gates (big GEMMs) ----
  k_embed<<<LSRC * NB, 256, 0, stream>>>(src, LSRC, 0, emb, srcX);
  const int Mfull = LSRC * NB;  // 4096
  k_gemm<<<gemm_blocks(Mfull, G4H), 256, 0, stream>>>(
      srcX, EDIM, eWihF0, EDIM, bihF0, bhhF0, pgA, G4H, nullptr, 0, Mfull, G4H, EDIM, 0);
  k_gemm<<<gemm_blocks(Mfull, G4H), 256, 0, stream>>>(
      srcX, EDIM, eWihB0, EDIM, bihB0, bhhB0, pgB, G4H, nullptr, 0, Mfull, G4H, EDIM, 0);

  // ---- 4. encoder layer 0: 128 fused recurrent steps, fwd+bwd in one launch ----
  for (int t = 0; t < LSRC; ++t) {
    StepArgs a = {};
    const int tb = LSRC - 1 - t;
    a.pg[0] = pgA + (long)t  * NB * G4H;  a.pg[1] = pgB + (long)tb * NB * G4H;
    a.Whh[0] = eWhhF0;                    a.Whh[1] = eWhhB0;
    a.hIn[0] = hF0[t & 1];  a.hOut[0] = hF0[(t + 1) & 1];
    a.hIn[1] = hB0[t & 1];  a.hOut[1] = hB0[(t + 1) & 1];
    a.c[0] = cF0;           a.c[1] = cB0;
    a.y[0] = y0 + (long)t  * NB * 2 * HDIM;
    a.y[1] = y0 + (long)tb * NB * 2 * HDIM;
    a.ldy = 2 * HDIM; a.yoff[0] = 0; a.yoff[1] = HDIM;
    k_lstm_step<<<dim3(HDIM / 16, 2), 256, 0, stream>>>(a);
  }

  // ---- 5. encoder layer 1 pre-gates (K = 2H), then 128 steps ----
  k_gemm<<<gemm_blocks(Mfull, G4H), 256, 0, stream>>>(
      y0, 2 * HDIM, eWihF1, 2 * HDIM, bihF1, bhhF1, pgA, G4H, nullptr, 0,
      Mfull, G4H, 2 * HDIM, 0);
  k_gemm<<<gemm_blocks(Mfull, G4H), 256, 0, stream>>>(
      y0, 2 * HDIM, eWihB1, 2 * HDIM, bihB1, bhhB1, pgB, G4H, nullptr, 0,
      Mfull, G4H, 2 * HDIM, 0);
  for (int t = 0; t < LSRC; ++t) {
    StepArgs a = {};
    const int tb = LSRC - 1 - t;
    a.pg[0] = pgA + (long)t  * NB * G4H;  a.pg[1] = pgB + (long)tb * NB * G4H;
    a.Whh[0] = eWhhF1;                    a.Whh[1] = eWhhB1;
    a.hIn[0] = hF1[t & 1];  a.hOut[0] = hF1[(t + 1) & 1];
    a.hIn[1] = hB1[t & 1];  a.hOut[1] = hB1[(t + 1) & 1];
    a.c[0] = cF1;           a.c[1] = cB1;
    a.y[0] = y1 + (long)t  * NB * 2 * HDIM;
    a.y[1] = y1 + (long)tb * NB * 2 * HDIM;
    a.ldy = 2 * HDIM; a.yoff[0] = 0; a.yoff[1] = HDIM;
    k_lstm_step<<<dim3(HDIM / 16, 2), 256, 0, stream>>>(a);
  }

  // ---- 6. attention projection: proj = enc_out @ att_W^T + att_b  (f16) ----
  k_gemm<<<gemm_blocks(Mfull, HDIM), 256, 0, stream>>>(
      y1, 2 * HDIM, attW, 2 * HDIM, att_b, nullptr, nullptr, 0, proj, HDIM,
      Mfull, HDIM, 2 * HDIM, 0);

  // ---- 7. decoder input embedding (teacher forcing -> all known) + pre-gates ----
  k_embed<<<LTGT * NB, 256, 0, stream>>>(tgt, LTGT, 1, emb, decX);
  const int Mdec = LTGT * NB;  // 1536
  k_gemm<<<gemm_blocks(Mdec, G4H), 256, 0, stream>>>(
      decX, EDIM, dWih0, EDIM, dbih0, dbhh0, decPG, G4H, nullptr, 0, Mdec, G4H, EDIM, 0);

  // ---- 8. decoder initial states: h = [enc_l0_fwd_final, enc_l0_bwd_final], c = init_c
  hipMemcpyAsync(dh0[0], hF0[0], (size_t)NB * HDIM * sizeof(half_t),
                 hipMemcpyDeviceToDevice, stream);
  hipMemcpyAsync(dh1[0], hB0[0], (size_t)NB * HDIM * sizeof(half_t),
                 hipMemcpyDeviceToDevice, stream);
  hipMemcpyAsync(dc0, init_c,             (size_t)NB * HDIM * sizeof(float),
                 hipMemcpyDeviceToDevice, stream);
  hipMemcpyAsync(dc1, init_c + NB * HDIM, (size_t)NB * HDIM * sizeof(float),
                 hipMemcpyDeviceToDevice, stream);

  // ---- 9. decoder: 48 steps ----
  for (int t = 0; t < LTGT; ++t) {
    const int p = t & 1;
    // layer 0 cell (pre-gates precomputed)
    {
      StepArgs a = {};
      a.pg[0]  = decPG + (long)t * NB * G4H;
      a.Whh[0] = dWhh0;
      a.hIn[0] = dh0[p]; a.hOut[0] = dh0[1 - p];
      a.c[0]   = dc0;
      k_lstm_step<<<dim3(HDIM / 16, 1), 256, 0, stream>>>(a);
    }
    // layer 1 cell (dynamic input h0 -> extra WMMA loop)
    {
      StepArgs a = {};
      a.x[0]  = dh0[1 - p]; a.Wx[0] = dWih1; a.Kx = HDIM;
      a.Whh[0] = dWhh1;
      a.hIn[0] = dh1[p]; a.hOut[0] = dh1[1 - p];
      a.c[0]   = dc1;
      a.b1[0]  = dbih1;  a.b2[0] = dbhh1;
      k_lstm_step<<<dim3(HDIM / 16, 1), 256, 0, stream>>>(a);
    }
    half_t* h1 = dh1[1 - p];
    // attention
    k_energy<<<(NB * LSRC) / 8, 256, 0, stream>>>(proj, h1, energy);
    k_softmax<<<NB, LSRC, 0, stream>>>(energy, score);
    k_ctx<<<NB, 256, 0, stream>>>(score, y1, h1, cat);
    // hid = tanh(cat @ lin1_W^T + b)
    k_gemm<<<gemm_blocks(NB, HDIM), 256, 0, stream>>>(
        cat, 3 * HDIM, lin1W, 3 * HDIM, lin1_b, nullptr, nullptr, 0, hid, HDIM,
        NB, HDIM, 3 * HDIM, 1);
    // logits = hid @ lin2_W^T + b  -> out[n][t][v]
    k_gemm<<<gemm_blocks(NB, VOCAB), 256, 0, stream>>>(
        hid, HDIM, lin2W, HDIM, lin2_b, nullptr,
        out + (long)t * VOCAB, (long)LTGT * VOCAB, nullptr, 0,
        NB, VOCAB, HDIM, 0);
  }
}
